// BabyGPT_21827023798615
// MI455X (gfx1250) — compile-verified, matrix operand order checked by reading, not run on hardware
//
#include <hip/hip_runtime.h>
#include <hip/hip_bf16.h>
#include <math.h>

// ---------------------------------------------------------------- constants
#define Vv 50257
#define Tt 1024
#define Ee 1024
#define Hh 16
#define Ll 12
#define Bb 4
#define HD 64
#define FFf 4096
#define BT (Bb * Tt)

typedef __attribute__((ext_vector_type(16))) __bf16 v16bf;
typedef __attribute__((ext_vector_type(8)))  float  v8f;

#define WMMA_BF16(a, b, c) \
  __builtin_amdgcn_wmma_f32_16x16x32_bf16(false, (a), false, (b), (short)0, (c), false, false)

// Fragment k-index for the documented 16-bit A/B 16x32 layout:
// vgpr v (=e>>1): base = (v&3)*2 + (v>>2)*16 ; element k = base + (e&1) + hi*8
__device__ __forceinline__ int frag_k(int e, int hi) {
  int v = e >> 1;
  return (v & 3) * 2 + (v >> 2) * 16 + (e & 1) + hi * 8;
}

// ---------------------------------------------------------------- f32 -> bf16
__global__ void f32_to_bf16_kernel(const float* __restrict__ in, __bf16* __restrict__ out, long n) {
  long i = (long)blockIdx.x * blockDim.x + threadIdx.x;
  if (i < n) out[i] = (__bf16)in[i];
}

// ---------------------------------------------------------------- embedding
__global__ void embed_kernel(const int* __restrict__ idx, const float* __restrict__ tok,
                             const float* __restrict__ pos, float* __restrict__ x) {
  int bt = blockIdx.x;
  int t  = bt % Tt;
  int id = idx[bt];
  const float* te = tok + (long)id * Ee;
  const float* pe = pos + (long)t  * Ee;
  float* xo = x + (long)bt * Ee;
  for (int e = threadIdx.x; e < Ee; e += blockDim.x) xo[e] = te[e] + pe[e];
}

// ---------------------------------------------------------------- layernorm (f32 in -> bf16 out)
__global__ void layernorm_kernel(const float* __restrict__ x, const float* __restrict__ g,
                                 const float* __restrict__ bta, __bf16* __restrict__ out) {
  __shared__ float s1[256];
  __shared__ float s2[256];
  int row = blockIdx.x;
  const float* xr = x + (long)row * Ee;
  float a = 0.f, q = 0.f;
  for (int e = threadIdx.x; e < Ee; e += 256) { float v = xr[e]; a += v; q += v * v; }
  s1[threadIdx.x] = a; s2[threadIdx.x] = q;
  __syncthreads();
  for (int s = 128; s > 0; s >>= 1) {
    if (threadIdx.x < s) { s1[threadIdx.x] += s1[threadIdx.x + s]; s2[threadIdx.x] += s2[threadIdx.x + s]; }
    __syncthreads();
  }
  float mean = s1[0] * (1.0f / Ee);
  float var  = s2[0] * (1.0f / Ee) - mean * mean;
  float rstd = rsqrtf(var + 1e-5f);
  __bf16* o = out + (long)row * Ee;
  for (int e = threadIdx.x; e < Ee; e += 256)
    o[e] = (__bf16)((xr[e] - mean) * rstd * g[e] + bta[e]);
}

// ---------------------------------------------------------------- bf16 WMMA GEMM
// C[M,N] = A[M,K] * B[K,N]  (A,B row-major bf16)
// Block 128x128, BK=32, 256 threads = 8 waves; wave tile 32x64 (2x4 of 16x16).
// A tile staged with GLOBAL_LOAD_ASYNC_TO_LDS_B128 (ASYNCcnt path, no VGPR copy);
// B tile staged through VGPRs because it is transposed into LDS.
// EPI: 0 = bf16 out; 1 = f32 out + bias + residual; 2 = bf16 out + bias + exact GELU;
//      3 = f32 out + bias.
#define GBM 128
#define GBN 128
#define GBK 32
#define GSA (GBK + 8)   // 40 bf16 = 80B rows: 16B-aligned for async B128, conflict-free reads
#define GSB (GBK + 4)   // 36 bf16 = 72B rows: conflict-free per-lane row reads

template <int EPI>
__global__ __launch_bounds__(256) void gemm_bf16_kernel(
    const __bf16* __restrict__ A, const __bf16* __restrict__ Bw,
    const float* __restrict__ bias, const float* res,
    float* outF, __bf16* outB, int M, int N, int K) {
  __shared__ __bf16 As[GBM * GSA];
  __shared__ __bf16 Bs[GBN * GSB];

  int tid = threadIdx.x;
  int wave = tid >> 5, lane = tid & 31;
  int lane16 = lane & 15, hi = lane >> 4;
  int wm = wave >> 1, wn = wave & 1;        // 4 wave-rows x 2 wave-cols
  int bn0 = blockIdx.x * GBN;
  int bm0 = blockIdx.y * GBM;

  // Per-thread async-copy geometry for the A tile: 512 x 16B chunks, 2 per thread.
  // chunk c: row m = c>>2, 16B-chunk-in-row cc = c&3 (row = 32 bf16 = 4 chunks).
  int c0 = tid * 2;
  int am0 = c0 >> 2, acc0 = c0 & 3;
  int am1 = (c0 + 1) >> 2, acc1 = (c0 + 1) & 3;
  // LDS byte addresses (low 32 bits of the generic pointer = LDS offset, ISA 10.2)
  unsigned lds_a0 = (unsigned)(uintptr_t)&As[am0 * GSA + acc0 * 8];
  unsigned lds_a1 = (unsigned)(uintptr_t)&As[am1 * GSA + acc1 * 8];

  v8f zero = {};
  v8f acc[2][4];
#pragma unroll
  for (int i = 0; i < 2; ++i)
#pragma unroll
    for (int j = 0; j < 4; ++j) acc[i][j] = zero;

  for (int k0 = 0; k0 < K; k0 += GBK) {
    // --- stage A tile 128x32 via async global->LDS B128 (GVS: saddr=A, voffset=bytes)
    {
      unsigned goff0 = (unsigned)(((long)(bm0 + am0) * K + k0 + acc0 * 8) * 2);
      unsigned goff1 = (unsigned)(((long)(bm0 + am1) * K + k0 + acc1 * 8) * 2);
      asm volatile("global_load_async_to_lds_b128 %0, %1, %2"
                   :: "v"(lds_a0), "v"(goff0), "s"(A) : "memory");
      asm volatile("global_load_async_to_lds_b128 %0, %1, %2"
                   :: "v"(lds_a1), "v"(goff1), "s"(A) : "memory");
    }
    // --- stage B tile 32x128, transposed into Bs[n][k] (N-guarded for head GEMM)
    {
      int base = tid * 16;
      int kk = base >> 7, n = base & 127;
      const __bf16* bp = Bw + (long)(k0 + kk) * N + bn0 + n;
#pragma unroll
      for (int i = 0; i < 16; ++i) {
        __bf16 vb = (__bf16)0.0f;
        if (bn0 + n + i < N) vb = bp[i];
        Bs[(n + i) * GSB + kk] = vb;
      }
      if (k0 + GBK < K) __builtin_prefetch(bp + (long)GBK * N, 0, 0);
    }
    // async A writes must land in LDS before any wave reads the tile
    asm volatile("s_wait_asynccnt 0x0" ::: "memory");
    __syncthreads();

    // --- fragments + 8 WMMAs
    v16bf afr[2], bfr[4];
#pragma unroll
    for (int i = 0; i < 2; ++i) {
      const __bf16* p = As + (wm * 32 + i * 16 + lane16) * GSA;
#pragma unroll
      for (int e = 0; e < 16; ++e) afr[i][e] = p[frag_k(e, hi)];
    }
#pragma unroll
    for (int j = 0; j < 4; ++j) {
      const __bf16* p = Bs + (wn * 64 + j * 16 + lane16) * GSB;
#pragma unroll
      for (int e = 0; e < 16; ++e) bfr[j][e] = p[frag_k(e, hi)];
    }
#pragma unroll
    for (int i = 0; i < 2; ++i)
#pragma unroll
      for (int j = 0; j < 4; ++j) acc[i][j] = WMMA_BF16(afr[i], bfr[j], acc[i][j]);
    __syncthreads();
  }

  // --- epilogue (C layout: n = lane%16 ; m = r + 8*hi)
#pragma unroll
  for (int i = 0; i < 2; ++i) {
#pragma unroll
    for (int j = 0; j < 4; ++j) {
      int n = bn0 + wn * 64 + j * 16 + lane16;
      if (n < N) {
        float bv = (EPI == 0) ? 0.0f : bias[n];
#pragma unroll
        for (int r = 0; r < 8; ++r) {
          int m = bm0 + wm * 32 + i * 16 + r + 8 * hi;
          long oi = (long)m * N + n;
          float v = acc[i][j][r];
          if (EPI == 0) {
            outB[oi] = (__bf16)v;
          } else if (EPI == 1) {
            outF[oi] = v + bv + res[oi];
          } else if (EPI == 2) {
            float t = v + bv;
            outB[oi] = (__bf16)(0.5f * t * (1.0f + erff(t * 0.70710678118654752f)));
          } else {
            outF[oi] = v + bv;
          }
        }
      }
    }
  }
}

// ---------------------------------------------------------------- attention
// One workgroup = (b, h, 64 query rows). Full 64x1024 f32 score matrix in LDS
// (320KB/WGP makes this possible -> no online-softmax rescaling).
#define SROW 1028  // padded row stride (floats): lanes m=0..15 -> distinct banks
#define ATT_LDS ((64 * SROW + 64) * sizeof(float))

__global__ __launch_bounds__(128) void attention_kernel(
    const __bf16* __restrict__ q, const __bf16* __restrict__ k,
    const __bf16* __restrict__ v, __bf16* __restrict__ att) {
  extern __shared__ float S[];          // [64][SROW] scores + [64] row sums
  float* Lrow = S + 64 * SROW;

  int q0 = blockIdx.x * 64;
  int h  = blockIdx.y;
  int b  = blockIdx.z;
  int tid = threadIdx.x;
  int wave = tid >> 5, lane = tid & 31;
  int lane16 = lane & 15, hi = lane >> 4;
  const float scale = 0.125f;           // HD^-0.5

  // ---- pass 1: S = mask(QK^T * scale), only k-tiles touching the causal region
  int kb_max = q0 / 16 + 4;
  int qrowA = q0 + wave * 16 + lane16;  // A-fragment row for this lane
  for (int kb = 0; kb < kb_max; ++kb) {
    v8f acc = {};
#pragma unroll
    for (int kk = 0; kk < HD; kk += 32) {
      v16bf a, bm;
      const __bf16* qp = q + ((long)(b * Tt + qrowA) * Hh + h) * HD + kk;
      const __bf16* kp = k + ((long)(b * Tt + kb * 16 + lane16) * Hh + h) * HD + kk;
#pragma unroll
      for (int e = 0; e < 16; ++e) {
        int ki = frag_k(e, hi);
        a[e]  = qp[ki];
        bm[e] = kp[ki];
      }
      acc = WMMA_BF16(a, bm, acc);
    }
    int kcol = kb * 16 + lane16;
#pragma unroll
    for (int r = 0; r < 8; ++r) {
      int lrow = wave * 16 + r + 8 * hi;
      float sv = acc[r] * scale;
      if (kcol > q0 + lrow) sv = -__builtin_inff();
      S[lrow * SROW + kcol] = sv;
    }
  }
  __syncthreads();

  // ---- softmax: lane pair (l, l+16) splits each row's columns, combine via shfl_xor
  {
    int lrow = wave * 16 + lane16;
    int ncol = kb_max * 16;
    int half = ncol >> 1;
    int cc0 = hi * half, cc1 = cc0 + half;
    float m = -__builtin_inff();
    for (int c = cc0; c < cc1; ++c) m = fmaxf(m, S[lrow * SROW + c]);
    m = fmaxf(m, __shfl_xor(m, 16, 32));
    float l = 0.f;
    for (int c = cc0; c < cc1; ++c) {
      float p = __expf(S[lrow * SROW + c] - m);
      S[lrow * SROW + c] = p;      // store unnormalized exp; scale by 1/l at the end
      l += p;
    }
    l += __shfl_xor(l, 16, 32);
    if (hi == 0) Lrow[lrow] = l;
  }
  __syncthreads();

  // ---- pass 2: O = P * V  (P read from LDS straight into bf16 A-fragments)
  v8f zero = {};
  v8f o[4];
#pragma unroll
  for (int j = 0; j < 4; ++j) o[j] = zero;

  int nkb = (q0 + 64) / 32;
  int m_local = wave * 16 + lane16;
  for (int kb = 0; kb < nkb; ++kb) {
    v16bf a;
#pragma unroll
    for (int e = 0; e < 16; ++e)
      a[e] = (__bf16)S[m_local * SROW + kb * 32 + frag_k(e, hi)];
#pragma unroll
    for (int tn = 0; tn < 4; ++tn) {
      v16bf bm;
      int d = tn * 16 + lane16;
#pragma unroll
      for (int e = 0; e < 16; ++e) {
        int key = kb * 32 + frag_k(e, hi);
        bm[e] = v[((long)(b * Tt + key) * Hh + h) * HD + d];
      }
      o[tn] = WMMA_BF16(a, bm, o[tn]);
    }
  }

  // ---- epilogue: scale rows by 1/l, store bf16 att[b,t,h,d]
#pragma unroll
  for (int tn = 0; tn < 4; ++tn) {
    int d = tn * 16 + lane16;
#pragma unroll
    for (int r = 0; r < 8; ++r) {
      int lrow = wave * 16 + r + 8 * hi;
      float inv_l = 1.0f / Lrow[lrow];
      att[((long)(b * Tt + q0 + lrow) * Hh + h) * HD + d] = (__bf16)(o[tn][r] * inv_l);
    }
  }
}

// ---------------------------------------------------------------- loss
__global__ void loss_row_kernel(const float* __restrict__ logits, const int* __restrict__ targets,
                                float* __restrict__ nll) {
  __shared__ float red[256];
  int row = blockIdx.x;
  const float* lr = logits + (long)row * Vv;
  float m = -__builtin_inff();
  for (int i = threadIdx.x; i < Vv; i += 256) m = fmaxf(m, lr[i]);
  red[threadIdx.x] = m; __syncthreads();
  for (int s = 128; s > 0; s >>= 1) {
    if (threadIdx.x < s) red[threadIdx.x] = fmaxf(red[threadIdx.x], red[threadIdx.x + s]);
    __syncthreads();
  }
  m = red[0]; __syncthreads();
  float sum = 0.f;
  for (int i = threadIdx.x; i < Vv; i += 256) sum += __expf(lr[i] - m);
  red[threadIdx.x] = sum; __syncthreads();
  for (int s = 128; s > 0; s >>= 1) {
    if (threadIdx.x < s) red[threadIdx.x] += red[threadIdx.x + s];
    __syncthreads();
  }
  if (threadIdx.x == 0) nll[row] = (m + __logf(red[0])) - lr[targets[row]];
}

__global__ void loss_mean_kernel(const float* __restrict__ nll, float* loss_out) {
  __shared__ float red[256];
  float s = 0.f;
  for (int i = threadIdx.x; i < BT; i += 256) s += nll[i];
  red[threadIdx.x] = s; __syncthreads();
  for (int st = 128; st > 0; st >>= 1) {
    if (threadIdx.x < st) red[threadIdx.x] += red[threadIdx.x + st];
    __syncthreads();
  }
  if (threadIdx.x == 0) *loss_out = red[0] * (1.0f / BT);
}

// ---------------------------------------------------------------- launch
extern "C" void kernel_launch(void* const* d_in, const int* in_sizes, int n_in,
                              void* d_out, int out_size, void* d_ws, size_t ws_size,
                              hipStream_t stream) {
  (void)in_sizes; (void)n_in; (void)out_size; (void)ws_size;

  const int*   idx     = (const int*)d_in[0];
  const int*   targets = (const int*)d_in[1];
  const float* tok_emb = (const float*)d_in[2];
  const float* pos_emb = (const float*)d_in[3];
  const float* Wq      = (const float*)d_in[4];
  const float* Wk      = (const float*)d_in[5];
  const float* Wv      = (const float*)d_in[6];
  const float* Wo      = (const float*)d_in[7];
  const float* bo      = (const float*)d_in[8];
  const float* ln1_g   = (const float*)d_in[9];
  const float* ln1_b   = (const float*)d_in[10];
  const float* ln2_g   = (const float*)d_in[11];
  const float* ln2_b   = (const float*)d_in[12];
  const float* W1      = (const float*)d_in[13];
  const float* b1      = (const float*)d_in[14];
  const float* W2      = (const float*)d_in[15];
  const float* b2      = (const float*)d_in[16];
  const float* lnf_g   = (const float*)d_in[17];
  const float* lnf_b   = (const float*)d_in[18];
  const float* Whead   = (const float*)d_in[19];
  const float* bhead   = (const float*)d_in[20];

  // ---- workspace layout
  size_t off = 0;
  auto alloc = [&](size_t bytes) -> char* {
    char* p = (char*)d_ws + off;
    off += (bytes + 255) & ~(size_t)255;
    return p;
  };
  __bf16* wq_b = (__bf16*)alloc((size_t)Ll * Ee * Ee * 2);
  __bf16* wk_b = (__bf16*)alloc((size_t)Ll * Ee * Ee * 2);
  __bf16* wv_b = (__bf16*)alloc((size_t)Ll * Ee * Ee * 2);
  __bf16* wo_b = (__bf16*)alloc((size_t)Ll * Ee * Ee * 2);
  __bf16* w1_b = (__bf16*)alloc((size_t)Ll * Ee * FFf * 2);
  __bf16* w2_b = (__bf16*)alloc((size_t)Ll * FFf * Ee * 2);
  __bf16* wh_b = (__bf16*)alloc((size_t)Ee * Vv * 2);
  float*  x    = (float*) alloc((size_t)BT * Ee * 4);
  __bf16* hb   = (__bf16*)alloc((size_t)BT * Ee * 2);
  __bf16* qb   = (__bf16*)alloc((size_t)BT * Ee * 2);
  __bf16* kb   = (__bf16*)alloc((size_t)BT * Ee * 2);
  __bf16* vb   = (__bf16*)alloc((size_t)BT * Ee * 2);
  __bf16* attb = (__bf16*)alloc((size_t)BT * Ee * 2);
  __bf16* ffb  = (__bf16*)alloc((size_t)BT * FFf * 2);
  float*  nll  = (float*) alloc((size_t)BT * 4);

  auto launch_conv = [&](const float* src, __bf16* dst, long n) {
    unsigned blocks = (unsigned)((n + 255) / 256);
    f32_to_bf16_kernel<<<blocks, 256, 0, stream>>>(src, dst, n);
  };

  // ---- one-time (per call) bf16 weight conversion
  launch_conv(Wq,    wq_b, (long)Ll * Ee * Ee);
  launch_conv(Wk,    wk_b, (long)Ll * Ee * Ee);
  launch_conv(Wv,    wv_b, (long)Ll * Ee * Ee);
  launch_conv(Wo,    wo_b, (long)Ll * Ee * Ee);
  launch_conv(W1,    w1_b, (long)Ll * Ee * FFf);
  launch_conv(W2,    w2_b, (long)Ll * FFf * Ee);
  launch_conv(Whead, wh_b, (long)Ee * Vv);

  // ---- embedding
  embed_kernel<<<BT, 256, 0, stream>>>(idx, tok_emb, pos_emb, x);

  dim3 gEE((Ee + GBN - 1) / GBN, BT / GBM);
  dim3 gEF((FFf + GBN - 1) / GBN, BT / GBM);

  // ---- transformer layers
  for (int l = 0; l < Ll; ++l) {
    layernorm_kernel<<<BT, 256, 0, stream>>>(x, ln1_g + (size_t)l * Ee, ln1_b + (size_t)l * Ee, hb);

    gemm_bf16_kernel<0><<<gEE, 256, 0, stream>>>(hb, wq_b + (size_t)l * Ee * Ee,
                                                 nullptr, nullptr, nullptr, qb, BT, Ee, Ee);
    gemm_bf16_kernel<0><<<gEE, 256, 0, stream>>>(hb, wk_b + (size_t)l * Ee * Ee,
                                                 nullptr, nullptr, nullptr, kb, BT, Ee, Ee);
    gemm_bf16_kernel<0><<<gEE, 256, 0, stream>>>(hb, wv_b + (size_t)l * Ee * Ee,
                                                 nullptr, nullptr, nullptr, vb, BT, Ee, Ee);

    attention_kernel<<<dim3(Tt / 64, Hh, Bb), 128, ATT_LDS, stream>>>(qb, kb, vb, attb);

    gemm_bf16_kernel<1><<<gEE, 256, 0, stream>>>(attb, wo_b + (size_t)l * Ee * Ee,
                                                 bo + (size_t)l * Ee, x, x, nullptr, BT, Ee, Ee);

    layernorm_kernel<<<BT, 256, 0, stream>>>(x, ln2_g + (size_t)l * Ee, ln2_b + (size_t)l * Ee, hb);

    gemm_bf16_kernel<2><<<gEF, 256, 0, stream>>>(hb, w1_b + (size_t)l * Ee * FFf,
                                                 b1 + (size_t)l * FFf, nullptr, nullptr, ffb,
                                                 BT, FFf, Ee);
    gemm_bf16_kernel<1><<<gEE, 256, 0, stream>>>(ffb, w2_b + (size_t)l * FFf * Ee,
                                                 b2 + (size_t)l * Ee, x, x, nullptr, BT, Ee, FFf);
  }

  // ---- final LN + head + loss
  layernorm_kernel<<<BT, 256, 0, stream>>>(x, lnf_g, lnf_b, hb);

  float* logits = (float*)d_out;
  dim3 gHead((Vv + GBN - 1) / GBN, BT / GBM);
  gemm_bf16_kernel<3><<<gHead, 256, 0, stream>>>(hb, wh_b, bhead, nullptr, logits, nullptr,
                                                 BT, Vv, Ee);

  loss_row_kernel<<<BT, 256, 0, stream>>>(logits, targets, nll);
  loss_mean_kernel<<<1, 256, 0, stream>>>(nll, logits + (size_t)BT * Vv);
}
